// MoEFFNLayer_5420248727733
// MI455X (gfx1250) — compile-verified, matrix operand order checked by reading, not run on hardware
//
#include <hip/hip_runtime.h>
#include <hip/hip_bf16.h>

#define H_DIM 1024
#define I_DIM 4096
#define NE 8
#define TT 8192
#define TILE_M 32
#define CHUNK 128
#define NTHREADS 512

typedef __attribute__((ext_vector_type(16))) __bf16 v16bf;
typedef __attribute__((ext_vector_type(8)))  float  v8f;

union ABFrag {
    v16bf v;
    uint4 q[2];
    unsigned int u32[8];
};

// Hardware packed convert: dst.lo = bf16(a), dst.hi = bf16(b)  (RNE)
__device__ __forceinline__ unsigned int cvt_pk_bf16(float a, float b) {
    unsigned int r;
    asm volatile("v_cvt_pk_bf16_f32 %0, %1, %2" : "=v"(r) : "v"(a), "v"(b));
    return r;
}

__device__ __forceinline__ unsigned short f2bf(float f) {
    return (unsigned short)(cvt_pk_bf16(f, 0.0f) & 0xFFFFu);
}

// Async global -> LDS copy of 16B per lane (ASYNCcnt-tracked, gfx1250).
// lptr: generic pointer to __shared__; low 32 bits carry the LDS offset.
__device__ __forceinline__ void async_load_b128(const void* gptr, void* lptr) {
    unsigned loff = (unsigned)(size_t)lptr;
    asm volatile("global_load_async_to_lds_b128 %0, %1, off"
                 :: "v"(loff), "v"(gptr) : "memory");
}
__device__ __forceinline__ void wait_asynccnt0() {
    asm volatile("s_wait_asynccnt 0x0" ::: "memory");
}

// A fragment (16x32 bf16) from LDS, row-major ushort array, leading dim ld.
// lane: m = lm; element i -> k = (i<8 ? 8*lh+i : 8+8*lh+i) + col0
__device__ __forceinline__ v16bf load_a_frag_lds(const unsigned short* Ls, int row0,
                                                 int ld, int col0, int lm, int lh) {
    const unsigned short* p = Ls + (row0 + lm) * ld + col0;
    ABFrag r;
    r.q[0] = *(const uint4*)(p + 8 * lh);
    r.q[1] = *(const uint4*)(p + 16 + 8 * lh);
    return r.v;
}

// B fragment (32x16): B(k,n) = W[(row0+n)*ldw + col0 + k], bf16 source.
__device__ __forceinline__ v16bf load_b_frag_bf16(const unsigned short* __restrict__ W,
                                                  int row0, int ldw, int col0,
                                                  int lm, int lh) {
    const unsigned short* p = W + (size_t)(row0 + lm) * ldw + col0;
    ABFrag r;
    r.q[0] = *(const uint4*)(p + 8 * lh);
    r.q[1] = *(const uint4*)(p + 16 + 8 * lh);
    return r.v;
}

// Fallback: fp32 source + packed hw convert (used only if ws too small)
__device__ __forceinline__ v16bf load_b_frag_f32(const float* __restrict__ W, int row0,
                                                 int ldw, int col0, int lm, int lh) {
    const float* p = W + (size_t)(row0 + lm) * ldw + col0;
    float4 a0 = *(const float4*)(p + 8 * lh);
    float4 a1 = *(const float4*)(p + 8 * lh + 4);
    float4 b0 = *(const float4*)(p + 16 + 8 * lh);
    float4 b1 = *(const float4*)(p + 16 + 8 * lh + 4);
    ABFrag r;
    r.u32[0] = cvt_pk_bf16(a0.x, a0.y);
    r.u32[1] = cvt_pk_bf16(a0.z, a0.w);
    r.u32[2] = cvt_pk_bf16(a1.x, a1.y);
    r.u32[3] = cvt_pk_bf16(a1.z, a1.w);
    r.u32[4] = cvt_pk_bf16(b0.x, b0.y);
    r.u32[5] = cvt_pk_bf16(b0.z, b0.w);
    r.u32[6] = cvt_pk_bf16(b1.x, b1.y);
    r.u32[7] = cvt_pk_bf16(b1.z, b1.w);
    return r.v;
}

#define WMMA_BF16(a, b, c) \
    __builtin_amdgcn_wmma_f32_16x16x32_bf16(false, (a), false, (b), (short)0, (c), false, false)

// ---------------------------------------------------------------------------
// fp32 -> bf16 bulk conversion (8 elements/thread, packed hw convert)
// ---------------------------------------------------------------------------
__global__ __launch_bounds__(256) void cvt_bf16_kernel(const float* __restrict__ src,
                                                       unsigned short* __restrict__ dst,
                                                       long n) {
    long i = ((long)blockIdx.x * 256 + threadIdx.x) * 8;
    if (i >= n) return;
    float4 a = *(const float4*)(src + i);
    float4 b = *(const float4*)(src + i + 4);
    uint4 o;
    o.x = cvt_pk_bf16(a.x, a.y);
    o.y = cvt_pk_bf16(a.z, a.w);
    o.z = cvt_pk_bf16(b.x, b.y);
    o.w = cvt_pk_bf16(b.z, b.w);
    *(uint4*)(dst + i) = o;
}

// ---------------------------------------------------------------------------
// Kernel 1: gating
// ---------------------------------------------------------------------------
__global__ __launch_bounds__(256) void gate_kernel(
    const float* __restrict__ x, const float* __restrict__ gw,
    float* __restrict__ probs, int* __restrict__ top_idx,
    float* __restrict__ top_w, int* __restrict__ usage) {
    __shared__ float gws[NE][H_DIM];
    int tid = threadIdx.x;
    for (int i = tid; i < NE * H_DIM; i += 256) gws[i / H_DIM][i % H_DIM] = gw[i];
    __syncthreads();

    int t = blockIdx.x * 256 + tid;
    float acc[NE];
#pragma unroll
    for (int e = 0; e < NE; e++) acc[e] = 0.0f;
    const float* xp = x + (size_t)t * H_DIM;
    for (int h = 0; h < H_DIM; h += 4) {
        float4 xv = *(const float4*)(xp + h);
#pragma unroll
        for (int e = 0; e < NE; e++)
            acc[e] += xv.x * gws[e][h] + xv.y * gws[e][h + 1] +
                      xv.z * gws[e][h + 2] + xv.w * gws[e][h + 3];
    }
    float mx = acc[0];
#pragma unroll
    for (int e = 1; e < NE; e++) mx = fmaxf(mx, acc[e]);
    float s = 0.0f;
#pragma unroll
    for (int e = 0; e < NE; e++) { acc[e] = __expf(acc[e] - mx); s += acc[e]; }
    float inv = 1.0f / s;
    float p[NE];
#pragma unroll
    for (int e = 0; e < NE; e++) { p[e] = acc[e] * inv; probs[(size_t)t * NE + e] = p[e]; }

    int b0 = 0; float v0 = p[0];
#pragma unroll
    for (int e = 1; e < NE; e++) if (p[e] > v0) { v0 = p[e]; b0 = e; }
    int b1 = -1; float v1 = -1.0f;
#pragma unroll
    for (int e = 0; e < NE; e++) if (e != b0 && p[e] > v1) { v1 = p[e]; b1 = e; }
    float denom = v0 + v1 + 1e-9f;
    top_idx[t * 2 + 0] = b0; top_idx[t * 2 + 1] = b1;
    top_w[t * 2 + 0] = v0 / denom; top_w[t * 2 + 1] = v1 / denom;
    atomicAdd(&usage[b0], 1);
    atomicAdd(&usage[b1], 1);
}

// ---------------------------------------------------------------------------
// Kernel 2: routing lists
// ---------------------------------------------------------------------------
__global__ __launch_bounds__(256) void route_kernel(
    const int* __restrict__ top_idx, const float* __restrict__ top_w,
    int* __restrict__ cnt, int* __restrict__ list_tok, float* __restrict__ list_w) {
    int t = blockIdx.x * 256 + threadIdx.x;
    if (t >= TT) return;
#pragma unroll
    for (int k = 0; k < 2; k++) {
        int e = top_idx[t * 2 + k];
        float w = top_w[t * 2 + k];
        int pos = atomicAdd(&cnt[e], 1);
        list_tok[e * TT + pos] = t;
        list_w[e * TT + pos] = w;
    }
}

// ---------------------------------------------------------------------------
// Kernel 3: fused expert FFN, bf16 WMMA 16x16x32.
// 512 threads = 16 waves per block of 32 routed tokens. 128-column chunks:
// each wave owns one 16x16 tile and computes g AND u (one A feeds 2 WMMA
// streams), silu*u in registers, double-buffered Hs -> 1 barrier per chunk.
// 64 acc VGPRs/wave -> low pressure, 2x more waves/SIMD for latency hiding.
// Xs is staged via async global->LDS (ASYNCcnt) on the PRECONV path.
// ---------------------------------------------------------------------------
template <bool PRECONV>
__global__ __launch_bounds__(NTHREADS) void expert_ffn_kernel(
    const float* __restrict__ x, const unsigned short* __restrict__ xbf,
    const float* __restrict__ gup, const unsigned short* __restrict__ gupbf,
    const float* __restrict__ dwn, const unsigned short* __restrict__ dwnbf,
    const int* __restrict__ cnt, const int* __restrict__ list_tok,
    const float* __restrict__ list_w, float* __restrict__ out) {
    int e = blockIdx.y;
    int m0 = blockIdx.x * TILE_M;
    int ne = cnt[e];
    if (m0 >= ne) return;

    __shared__ unsigned short Xs[TILE_M][H_DIM];      // 64KB bf16 activations
    __shared__ unsigned short Hs[2][TILE_M][CHUNK];   // 16KB double-buffered silu(g)*u
    __shared__ int toks[TILE_M];
    __shared__ float wts[TILE_M];

    int tid = threadIdx.x;
    int wave = tid >> 5, lane = tid & 31, lm = lane & 15, lh = lane >> 4;

    if (tid < TILE_M) {
        int idx = m0 + tid;
        if (idx < ne) { toks[tid] = list_tok[e * TT + idx]; wts[tid] = list_w[e * TT + idx]; }
        else          { toks[tid] = list_tok[e * TT + ne - 1]; wts[tid] = 0.0f; }
    }
    __syncthreads();

    if (PRECONV) {
        // async global->LDS staging: 16B per lane per op, 8 ops per thread
        for (int idx = tid; idx < TILE_M * (H_DIM / 8); idx += NTHREADS) {
            int r = idx >> 7;          // H/8 == 128 chunks per row
            int c8 = idx & 127;
            async_load_b128(xbf + (size_t)toks[r] * H_DIM + 8 * c8, &Xs[r][8 * c8]);
        }
        wait_asynccnt0();
    } else {
        for (int idx = tid; idx < TILE_M * (H_DIM / 4); idx += NTHREADS) {
            int r = idx >> 8;
            int c4 = idx & 255;
            float4 v = *(const float4*)(x + (size_t)toks[r] * H_DIM + 4 * c4);
            uint2 sv;
            sv.x = cvt_pk_bf16(v.x, v.y);
            sv.y = cvt_pk_bf16(v.z, v.w);
            *(uint2*)(&Xs[r][4 * c4]) = sv;
        }
    }
    __syncthreads();

    // GEMM1 roles: 16 tiles = 2 M-tiles x 8 N-tiles, each wave owns one (g AND u)
    int mt1 = wave >> 3;
    int nt1 = wave & 7;
    // GEMM2 roles: wave -> (M-tile, 128-wide N range)
    int mt2 = wave >> 3;
    int nqq = wave & 7;

    const v8f vzero = (v8f){0.f, 0.f, 0.f, 0.f, 0.f, 0.f, 0.f, 0.f};
    v8f acc2[8];
#pragma unroll
    for (int i = 0; i < 8; i++) acc2[i] = vzero;

    const float* Wg           = gup   + (size_t)e * (2 * I_DIM) * H_DIM;
    const unsigned short* Wgb = gupbf + (size_t)e * (2 * I_DIM) * H_DIM;
    const float* Wd           = dwn   + (size_t)e * H_DIM * I_DIM;
    const unsigned short* Wdb = dwnbf + (size_t)e * H_DIM * I_DIM;

#define LOAD_B1(row, col) (PRECONV ? load_b_frag_bf16(Wgb, (row), H_DIM, (col), lm, lh) \
                                   : load_b_frag_f32 (Wg,  (row), H_DIM, (col), lm, lh))
#define LOAD_B2(row, col) (PRECONV ? load_b_frag_bf16(Wdb, (row), I_DIM, (col), lm, lh) \
                                   : load_b_frag_f32 (Wd,  (row), I_DIM, (col), lm, lh))

    int p = 0;
    for (int jc = 0; jc < I_DIM; jc += CHUNK) {
        // ---- GEMM1: g-tile and u-tile for this wave, K = 1024, pipelined
        int grow0 = jc + 16 * nt1;
        int urow0 = I_DIM + grow0;
        v8f accg = vzero, accu = vzero;
        v16bf a_c  = load_a_frag_lds(&Xs[0][0], 16 * mt1, H_DIM, 0, lm, lh);
        v16bf bg_c = LOAD_B1(grow0, 0);
        v16bf bu_c = LOAD_B1(urow0, 0);
#pragma unroll 4
        for (int kc = 0; kc < H_DIM - 32; kc += 32) {
            v16bf a_n  = load_a_frag_lds(&Xs[0][0], 16 * mt1, H_DIM, kc + 32, lm, lh);
            v16bf bg_n = LOAD_B1(grow0, kc + 32);
            v16bf bu_n = LOAD_B1(urow0, kc + 32);
            accg = WMMA_BF16(a_c, bg_c, accg);
            accu = WMMA_BF16(a_c, bu_c, accu);
            a_c = a_n; bg_c = bg_n; bu_c = bu_n;
        }
        accg = WMMA_BF16(a_c, bg_c, accg);
        accu = WMMA_BF16(a_c, bu_c, accu);

        // silu(g) * u locally, store bf16 into Hs[p]
#pragma unroll
        for (int r = 0; r < 8; r++) {
            int mrow = 16 * mt1 + r + 8 * lh;
            int ncol = 16 * nt1 + lm;
            float g = accg[r];
            float u = accu[r];
            Hs[p][mrow][ncol] = f2bf((g / (1.0f + __expf(-g))) * u);
        }
        __syncthreads();   // Hs[p] ready for all waves

        // ---- GEMM2: Y[32][1024] += Hs[p](32x128) @ Wd[:, jc:jc+128]^T, pipelined
        v16bf a2f[4];
#pragma unroll
        for (int kk = 0; kk < 4; kk++)
            a2f[kk] = load_a_frag_lds(&Hs[p][0][0], 16 * mt2, CHUNK, 32 * kk, lm, lh);
        v16bf b_c = LOAD_B2(128 * nqq, jc);
#pragma unroll
        for (int s = 0; s < 32; s++) {
            int t  = s >> 2, kk = s & 3;
            int tn = (s + 1) >> 2, kn = (s + 1) & 3;
            v16bf b_n = (s < 31) ? LOAD_B2(128 * nqq + 16 * tn, jc + 32 * kn) : b_c;
            acc2[t] = WMMA_BF16(a2f[kk], b_c, acc2[t]);
            b_c = b_n;
        }
        p ^= 1;
    }
#undef LOAD_B1
#undef LOAD_B2

    // Epilogue: weighted scatter-add (exactly 2 fp contributions per element)
#pragma unroll
    for (int t = 0; t < 8; t++) {
        int ncol = 128 * nqq + 16 * t + lm;
#pragma unroll
        for (int r = 0; r < 8; r++) {
            int mrow = 16 * mt2 + r + 8 * lh;
            float v = acc2[t][r] * wts[mrow];
            atomicAdd(out + (size_t)toks[mrow] * H_DIM + ncol, v);
        }
    }
}

// ---------------------------------------------------------------------------
// Kernel 4: aux loss — deterministic fixed-order reduction
// ---------------------------------------------------------------------------
__global__ __launch_bounds__(256) void aux_kernel(const float* __restrict__ probs,
                                                  const int* __restrict__ usage,
                                                  float* __restrict__ out_aux) {
    __shared__ float red[256];
    float local[NE];
#pragma unroll
    for (int e = 0; e < NE; e++) local[e] = 0.0f;
    for (int t = threadIdx.x; t < TT; t += 256) {
#pragma unroll
        for (int e = 0; e < NE; e++) local[e] += probs[(size_t)t * NE + e];
    }
    float total = 0.0f;
    for (int e = 0; e < NE; e++) {
        red[threadIdx.x] = local[e];
        __syncthreads();
        for (int s = 128; s > 0; s >>= 1) {
            if (threadIdx.x < (unsigned)s) red[threadIdx.x] += red[threadIdx.x + s];
            __syncthreads();
        }
        if (threadIdx.x == 0) {
            float imp = red[0] / (float)TT;
            float us = (float)usage[e] / ((float)TT * 2.0f + 1e-9f);
            total += us * imp;
        }
        __syncthreads();
    }
    if (threadIdx.x == 0) *out_aux = fminf(total * (float)NE * 0.01f, 1.0f);
}

// ---------------------------------------------------------------------------
extern "C" void kernel_launch(void* const* d_in, const int* in_sizes, int n_in,
                              void* d_out, int out_size, void* d_ws, size_t ws_size,
                              hipStream_t stream) {
    const float* x    = (const float*)d_in[0];
    const float* gw   = (const float*)d_in[1];
    const float* gup  = (const float*)d_in[2];
    const float* dwn  = (const float*)d_in[3];
    float* out = (float*)d_out;

    char* ws = (char*)d_ws;
    float* probs    = (float*)(ws + 0);        // T*8*4   = 262144
    int*   top_idx  = (int*)  (ws + 262144);   // 65536
    float* top_w    = (float*)(ws + 327680);   // 65536
    int*   cnt      = (int*)  (ws + 393216);   // 8 ints (256B pad)
    int*   usage    = (int*)  (ws + 393472);   // 8 ints (256B pad)
    int*   list_tok = (int*)  (ws + 393728);   // 262144
    float* list_w   = (float*)(ws + 655872);   // 262144

    const long n_x   = (long)TT * H_DIM;             //  8,388,608
    const long n_gup = (long)NE * 2 * I_DIM * H_DIM; // 67,108,864
    const long n_dwn = (long)NE * H_DIM * I_DIM;     // 33,554,432
    const size_t off_xbf   = 1048576;
    const size_t off_gupbf = off_xbf + (size_t)n_x * 2;
    const size_t off_dwnbf = off_gupbf + (size_t)n_gup * 2;
    const size_t ws_needed = off_dwnbf + (size_t)n_dwn * 2;  // ~219 MB
    unsigned short* xbf   = (unsigned short*)(ws + off_xbf);
    unsigned short* gupbf = (unsigned short*)(ws + off_gupbf);
    unsigned short* dwnbf = (unsigned short*)(ws + off_dwnbf);
    const bool preconv = (ws_size >= ws_needed);

    hipMemsetAsync(d_out, 0, (size_t)out_size * sizeof(float), stream);
    hipMemsetAsync(cnt, 0, 512, stream);  // cnt + usage

    if (preconv) {
        cvt_bf16_kernel<<<(int)(n_x / 8 / 256), 256, 0, stream>>>(x, xbf, n_x);
        cvt_bf16_kernel<<<(int)(n_gup / 8 / 256), 256, 0, stream>>>(gup, gupbf, n_gup);
        cvt_bf16_kernel<<<(int)(n_dwn / 8 / 256), 256, 0, stream>>>(dwn, dwnbf, n_dwn);
    }

    gate_kernel<<<TT / 256, 256, 0, stream>>>(x, gw, probs, top_idx, top_w, usage);
    route_kernel<<<TT / 256, 256, 0, stream>>>(top_idx, top_w, cnt, list_tok, list_w);
    if (preconv) {
        expert_ffn_kernel<true><<<dim3(TT / TILE_M, NE), NTHREADS, 0, stream>>>(
            x, xbf, gup, gupbf, dwn, dwnbf, cnt, list_tok, list_w, out);
    } else {
        expert_ffn_kernel<false><<<dim3(TT / TILE_M, NE), NTHREADS, 0, stream>>>(
            x, xbf, gup, gupbf, dwn, dwnbf, cnt, list_tok, list_w, out);
    }
    aux_kernel<<<1, 256, 0, stream>>>(probs, usage, out + (size_t)TT * H_DIM);
}
